// Encoder_35682588295656
// MI455X (gfx1250) — compile-verified, matrix operand order checked by reading, not run on hardware
//
#include <hip/hip_runtime.h>
#include <hip/hip_bf16.h>

typedef __attribute__((ext_vector_type(16))) _Float16 v16h;
typedef __attribute__((ext_vector_type(8)))  float    v8f;
typedef __attribute__((ext_vector_type(4)))  unsigned int v4u;
typedef __attribute__((ext_vector_type(8)))  int      v8i;
typedef __attribute__((ext_vector_type(4)))  int      v4i;

#define B_     32
#define TTOT_  48
#define T_     24
#define N_     1000
#define F_     16
#define HID_   64
#define EMB_   8
#define HEADS_ 4
#define DEG_   16
#define C_     17      // F+1
#define FCIN_  89      // 8 + 64 + 16 + 1
#define LDA_   96      // A stride in halves (even -> dword-aligned half2 pairs)
#define LDB_   96      // Bt stride in halves
#define WSLOT_ (64 * LDB_)   // halves per prepped weight slot (64 x 96 f16 tile)

// ---------------------------------------------------------------------------
// Fragment loader for V_WMMA_F32_16X16X32_F16.
// 16-bit A-matrix layout (ISA 7.12.2): lane l holds row M=l&15; lane group
// g=l>>4 adds 8 to K; vector elem pair v covers K = 2(v&3) + 16*(v>>2) + 8g.
// Bt is stored transposed (n-major) so the same loader serves both operands.
// ---------------------------------------------------------------------------
__device__ __forceinline__ v16h ld_frag(const _Float16* base, int ld,
                                        int tile, int kt, int lane)
{
    const int row   = tile * 16 + (lane & 15);
    const int kbase = kt * 32 + ((lane >> 4) << 3);
    union { v16h v; _Float16 h[16]; } u;
#pragma unroll
    for (int r = 0; r < 8; ++r) {
        const int k = kbase + ((r & 4) << 2) + ((r & 3) << 1);
        u.h[2 * r]     = base[row * ld + k];
        u.h[2 * r + 1] = base[row * ld + k + 1];
    }
    return u.v;
}

// 32x64 <- (32 x 32*ktiles) x (32*ktiles x 64), 8 waves -> one 16x16 tile each
__device__ __forceinline__ void gemm32x64(const _Float16* A, const _Float16* Bt,
                                          int ktiles, float* Cm, int lane, int wave)
{
    const int it = wave >> 2;          // row tile 0..1
    const int jt = wave & 3;           // col tile 0..3
    v8f acc = {};
    for (int kt = 0; kt < ktiles; ++kt) {
        v16h a = ld_frag(A,  LDA_, it, kt, lane);
        v16h b = ld_frag(Bt, LDB_, jt, kt, lane);
        acc = __builtin_amdgcn_wmma_f32_16x16x32_f16(false, a, false, b,
                                                     (short)0, acc, false, false);
    }
    // f32 C/D layout: VGPR r -> M = 16*it + r + 8*(lane>>4), N = 16*jt + (lane&15)
    const int col   = jt * 16 + (lane & 15);
    const int rbase = it * 16 + ((lane >> 4) << 3);
#pragma unroll
    for (int r = 0; r < 8; ++r)
        Cm[(rbase + r) * HID_ + col] = acc[r];
}

// Fallback: store W (K x 64, row-major f32) transposed into Bt[n][koff+k] f16
__device__ __forceinline__ void load_Bt(const float* W, int K, int koff,
                                        _Float16* Bt, int tid)
{
    for (int idx = tid; idx < K * HID_; idx += 256) {
        const int k = idx >> 6, nc = idx & 63;
        Bt[nc * LDB_ + koff + k] = (_Float16)W[idx];
    }
}

// ---------------------------------------------------------------------------
// Tensor Data Mover: DMA one prepped 64x96 f16 tile (contiguous 2D tensor,
// row = 96 halves) from global into LDS.  D# per CDNA5 ISA:
//   group0: count=1 | lds_addr | global_addr[56:0] | type=2
//   group1: data_size=2B, tensor_dim0=96, tensor_dim1=64, tile_dim0=96,
//           tile_dim1=64, tensor_dim0_stride=96
//   groups 2/3 (+spare group): zero (2D tile).  clang-23 6-arg builtin form.
// ---------------------------------------------------------------------------
__device__ __forceinline__ void tdm_load_Bt(const _Float16* gsrc, _Float16* ldsDst)
{
    const unsigned long long ga = (unsigned long long)(uintptr_t)gsrc;
    const unsigned int lo = (unsigned int)(uintptr_t)ldsDst;   // LDS byte offset
    v4u g0 = { 1u,                                   // count=1, user desc
               lo,                                   // lds_addr
               (unsigned int)ga,                     // global_addr[31:0]
               (unsigned int)(ga >> 32) | (2u << 30) // global_addr[56:32] | type=2
             };
    v8i g1 = { (1 << 16),          // [17:16] data_size = 1 -> 2 bytes
               (LDB_ << 16),       // [63:48] tensor_dim0[15:0] = 96
               (64  << 16),        // [95:80] tensor_dim1[15:0] = 64
               (LDB_ << 16),       // [127:112] tile_dim0 = 96
               64,                 // [143:128] tile_dim1 = 64 ; tile_dim2 = 0
               LDB_,               // [191:160] tensor_dim0_stride = 96
               0, 0 };
    v4i z4 = { 0, 0, 0, 0 };
    v8i z8 = { 0, 0, 0, 0, 0, 0, 0, 0 };
    __builtin_amdgcn_tensor_load_to_lds(g0, g1, z4, z4, z8, 0);
}

// ---------------------------------------------------------------------------
// One-time weight prep: seven pre-transposed, zero-padded 64x96 f16 B tiles.
//  slot 0: [W_rel ; W_root] (k 0..16 / 17..33)   slot 1: W_fc (k 0..88)
//  slots 2..6: Wq, Wk, Wv, Wo, Wmlp (k 0..63)
// ---------------------------------------------------------------------------
__global__ void prep_weights(const float* __restrict__ Wrel, const float* __restrict__ Wroot,
                             const float* __restrict__ Wfc,
                             const float* __restrict__ Wq, const float* __restrict__ Wk,
                             const float* __restrict__ Wv, const float* __restrict__ Wo,
                             const float* __restrict__ Wmlp, _Float16* __restrict__ ws)
{
    const int idx = blockIdx.x * 256 + threadIdx.x;
    if (idx >= 7 * WSLOT_) return;
    const int slot = idx / WSLOT_;
    const int r    = idx - slot * WSLOT_;
    const int nc   = r / LDB_;          // output column n (0..63)
    const int k    = r - nc * LDB_;     // K index (0..95)
    float v = 0.0f;
    switch (slot) {
        case 0: if (k < C_) v = Wrel[k * HID_ + nc];
                else if (k < 2 * C_) v = Wroot[(k - C_) * HID_ + nc]; break;
        case 1: if (k < FCIN_) v = Wfc[k * HID_ + nc]; break;
        case 2: if (k < HID_)  v = Wq[k * HID_ + nc];  break;
        case 3: if (k < HID_)  v = Wk[k * HID_ + nc];  break;
        case 4: if (k < HID_)  v = Wv[k * HID_ + nc];  break;
        case 5: if (k < HID_)  v = Wo[k * HID_ + nc];  break;
        default:if (k < HID_)  v = Wmlp[k * HID_ + nc]; break;
    }
    ws[idx] = (_Float16)v;
}

template <int USE_TDM>
__global__ __launch_bounds__(256)
void encoder_node_kernel(const float* __restrict__ Xg, const float* __restrict__ Yg,
                         const int*   __restrict__ esrc, const float* __restrict__ ew,
                         const float* __restrict__ pos,
                         const float* __restrict__ Wrel, const float* __restrict__ brel,
                         const float* __restrict__ Wroot,
                         const float* __restrict__ Wfc,  const float* __restrict__ bfc,
                         const float* __restrict__ Wq,   const float* __restrict__ bq,
                         const float* __restrict__ Wk,   const float* __restrict__ bk,
                         const float* __restrict__ Wv,   const float* __restrict__ bv,
                         const float* __restrict__ Wo,   const float* __restrict__ bo,
                         const float* __restrict__ Wmlp, const float* __restrict__ bmlp,
                         const _Float16* __restrict__ wsB,
                         float* __restrict__ out)
{
    __shared__ __align__(16) _Float16 smA[32 * LDA_];   // GEMM A operand
    __shared__ __align__(16) _Float16 smB[64 * LDB_];   // GEMM B operand (transposed)
    __shared__ __align__(16) float    smC[32 * HID_];   // GEMM output staging
    __shared__ float smX[T_ * 20];                       // self features [y | X]
    __shared__ float smQ[T_ * HID_];
    __shared__ float smK[T_ * HID_];
    __shared__ float smV[T_ * HID_];
    __shared__ int   smSrc[DEG_];
    __shared__ float smW[DEG_];

    const int tid  = threadIdx.x;
    const int lane = tid & 31;
    const int wave = tid >> 5;
    const int bn   = blockIdx.x;
    const int b    = bn / N_;
    const int n    = bn - b * N_;

    // ---- kick off DMA of B for stage 1 ASAP (overlaps aggregation) ----
    if (USE_TDM) { if (wave == 0) tdm_load_Bt(wsB + 0 * WSLOT_, smB); }

    if (tid < DEG_) {
        const int eb = (b * N_ + n) * DEG_;
        smSrc[tid] = esrc[eb + tid] - b * N_;   // local neighbor id in [0,N)
        smW[tid]   = ew[eb + tid];
    }
    {   // one-time zero of padded A buffer (B padding comes pre-zeroed via TDM)
        uint32_t* a32 = (uint32_t*)smA;
        for (int i = tid; i < (32 * LDA_) / 2; i += 256) a32[i] = 0u;
        if (!USE_TDM) {
            uint32_t* b32 = (uint32_t*)smB;
            for (int i = tid; i < (64 * LDB_) / 2; i += 256) b32[i] = 0u;
        }
    }
    __syncthreads();

    // ================= stage 1: graph aggregation + [agg|x] build ===========
    for (int idx = tid; idx < T_ * C_; idx += 256) {
        const int t = idx / C_, c = idx - t * C_;
        const int tb = (b * TTOT_ + t) * N_;
        const float selfv = (c == 0) ? Yg[tb + n] : Xg[(tb + n) * F_ + (c - 1)];
        float s = 0.0f;
#pragma unroll
        for (int e = 0; e < DEG_; ++e) {
            const int sn = smSrc[e];
            const float xv = (c == 0) ? Yg[tb + sn] : Xg[(tb + sn) * F_ + (c - 1)];
            s += smW[e] * xv;
        }
        smA[t * LDA_ + c]      = (_Float16)s;
        smA[t * LDA_ + C_ + c] = (_Float16)selfv;
        smX[t * 20 + c]        = selfv;
    }
    if (!USE_TDM) { load_Bt(Wrel, C_, 0, smB, tid); load_Bt(Wroot, C_, C_, smB, tid); }
    if (USE_TDM) __builtin_amdgcn_s_wait_tensorcnt(0);
    __syncthreads();

    gemm32x64(smA, smB, 2, smC, lane, wave);        // K = 64 (34 real + zeros)
    __syncthreads();

    // ================= stage 2: FC input build ==============================
    if (USE_TDM) { if (wave == 0) tdm_load_Bt(wsB + 1 * WSLOT_, smB); }
    for (int idx = tid; idx < T_ * FCIN_; idx += 256) {
        const int t = idx / FCIN_, c = idx - t * FCIN_;
        float vv;
        if (c < EMB_) {
            vv = pos[t * EMB_ + c];
        } else if (c < EMB_ + HID_) {
            const int cc = c - EMB_;
            const float pre = smC[t * HID_ + cc] + brel[cc];
            vv = 1.0f / (1.0f + __expf(-pre));
        } else if (c < EMB_ + HID_ + F_) {
            vv = smX[t * 20 + 1 + (c - EMB_ - HID_)];
        } else {
            vv = smX[t * 20];                       // y
        }
        smA[t * LDA_ + c] = (_Float16)vv;
    }
    if (!USE_TDM) load_Bt(Wfc, FCIN_, 0, smB, tid);
    if (USE_TDM) __builtin_amdgcn_s_wait_tensorcnt(0);
    __syncthreads();

    gemm32x64(smA, smB, 3, smC, lane, wave);        // K = 96 (89 real + zeros)
    __syncthreads();

    // z = smC + b_fc  (f16 A for q/k/v GEMMs)
    if (USE_TDM) { if (wave == 0) tdm_load_Bt(wsB + 2 * WSLOT_, smB); }
    for (int idx = tid; idx < T_ * HID_; idx += 256) {
        const int t = idx >> 6, c = idx & 63;
        smA[t * LDA_ + c] = (_Float16)(smC[idx] + bfc[c]);
    }
    if (!USE_TDM) load_Bt(Wq, HID_, 0, smB, tid);
    if (USE_TDM) __builtin_amdgcn_s_wait_tensorcnt(0);
    __syncthreads();

    gemm32x64(smA, smB, 2, smC, lane, wave);        // q
    __syncthreads();
    if (USE_TDM) { if (wave == 0) tdm_load_Bt(wsB + 3 * WSLOT_, smB); }
    for (int idx = tid; idx < T_ * HID_; idx += 256) {
        const int t = idx >> 6, c = idx & 63;
        smQ[idx] = (smC[idx] + bq[c]) * 0.25f;      // 1/sqrt(d), d=16
    }
    if (!USE_TDM) load_Bt(Wk, HID_, 0, smB, tid);
    if (USE_TDM) __builtin_amdgcn_s_wait_tensorcnt(0);
    __syncthreads();

    gemm32x64(smA, smB, 2, smC, lane, wave);        // k
    __syncthreads();
    if (USE_TDM) { if (wave == 0) tdm_load_Bt(wsB + 4 * WSLOT_, smB); }
    for (int idx = tid; idx < T_ * HID_; idx += 256) {
        const int t = idx >> 6, c = idx & 63;
        smK[idx] = smC[idx] + bk[c];
    }
    if (!USE_TDM) load_Bt(Wv, HID_, 0, smB, tid);
    if (USE_TDM) __builtin_amdgcn_s_wait_tensorcnt(0);
    __syncthreads();

    gemm32x64(smA, smB, 2, smC, lane, wave);        // v
    __syncthreads();
    if (USE_TDM) { if (wave == 0) tdm_load_Bt(wsB + 5 * WSLOT_, smB); }
    for (int idx = tid; idx < T_ * HID_; idx += 256) {
        const int t = idx >> 6, c = idx & 63;
        smV[idx] = smC[idx] + bv[c];
    }
    if (!USE_TDM) load_Bt(Wo, HID_, 0, smB, tid);
    __syncthreads();

    // ================= attention: 4 heads x 24 queries (96 lanes, f32) ======
    if (tid < HEADS_ * T_) {
        const int hh = tid / T_;
        const int i  = tid - hh * T_;
        const float* qrow = &smQ[i * HID_ + hh * 16];
        float sc[T_];
        float m = -3.0e38f;
        for (int j = 0; j < T_; ++j) {
            const float* krow = &smK[j * HID_ + hh * 16];
            float s = 0.0f;
#pragma unroll
            for (int d = 0; d < 16; ++d) s += qrow[d] * krow[d];
            sc[j] = s;
            m = fmaxf(m, s);
        }
        float sum = 0.0f;
        for (int j = 0; j < T_; ++j) { sc[j] = __expf(sc[j] - m); sum += sc[j]; }
        const float inv = 1.0f / sum;
#pragma unroll
        for (int d = 0; d < 16; ++d) {
            float cx = 0.0f;
            for (int j = 0; j < T_; ++j) cx += sc[j] * smV[j * HID_ + hh * 16 + d];
            smA[i * LDA_ + hh * 16 + d] = (_Float16)(cx * inv);   // ctx as next A
        }
    }
    if (USE_TDM) __builtin_amdgcn_s_wait_tensorcnt(0);   // Wo DMA (issued 2 phases ago)
    __syncthreads();

    gemm32x64(smA, smB, 2, smC, lane, wave);        // ctx @ Wo
    __syncthreads();
    if (USE_TDM) { if (wave == 0) tdm_load_Bt(wsB + 6 * WSLOT_, smB); }
    for (int idx = tid; idx < T_ * HID_; idx += 256) {
        const int t = idx >> 6, c = idx & 63;
        smA[t * LDA_ + c] = (_Float16)(smC[idx] + bo[c]);
    }
    if (!USE_TDM) load_Bt(Wmlp, HID_, 0, smB, tid);
    if (USE_TDM) __builtin_amdgcn_s_wait_tensorcnt(0);
    __syncthreads();

    gemm32x64(smA, smB, 2, smC, lane, wave);        // x_attn @ W_mlp
    __syncthreads();

    // ================= final write: out[b][t][n][c], 128-bit stores =========
    for (int idx = tid; idx < T_ * (HID_ / 4); idx += 256) {
        const int t  = idx >> 4;
        const int c4 = (idx & 15) * 4;
        float4 v;
        v.x = smC[t * HID_ + c4 + 0] + bmlp[c4 + 0];
        v.y = smC[t * HID_ + c4 + 1] + bmlp[c4 + 1];
        v.z = smC[t * HID_ + c4 + 2] + bmlp[c4 + 2];
        v.w = smC[t * HID_ + c4 + 3] + bmlp[c4 + 3];
        ((float4*)out)[((b * T_ + t) * N_ + n) * (HID_ / 4) + (idx & 15)] = v;
    }
}

extern "C" void kernel_launch(void* const* d_in, const int* in_sizes, int n_in,
                              void* d_out, int out_size, void* d_ws, size_t ws_size,
                              hipStream_t stream)
{
    (void)in_sizes; (void)n_in; (void)out_size;
    const float* X    = (const float*)d_in[0];
    const float* Y    = (const float*)d_in[1];
    const int*   esrc = (const int*)  d_in[2];
    /* d_in[3] = edge_dst: implicit (DEG contiguous edges per dst) */
    const float* ew   = (const float*)d_in[4];
    const float* pos  = (const float*)d_in[5];
    const float* Wrel = (const float*)d_in[6];
    const float* brel = (const float*)d_in[7];
    const float* Wroot= (const float*)d_in[8];
    const float* Wfc  = (const float*)d_in[9];
    const float* bfc  = (const float*)d_in[10];
    const float* Wq   = (const float*)d_in[11];
    const float* bq   = (const float*)d_in[12];
    const float* Wk   = (const float*)d_in[13];
    const float* bk   = (const float*)d_in[14];
    const float* Wv   = (const float*)d_in[15];
    const float* bv   = (const float*)d_in[16];
    const float* Wo   = (const float*)d_in[17];
    const float* bo   = (const float*)d_in[18];
    const float* Wmlp = (const float*)d_in[19];
    const float* bmlp = (const float*)d_in[20];
    float* out = (float*)d_out;

    const size_t wsNeed = (size_t)7 * WSLOT_ * sizeof(_Float16);
    if (ws_size >= wsNeed) {
        _Float16* wsB = (_Float16*)d_ws;
        prep_weights<<<dim3((7 * WSLOT_ + 255) / 256), dim3(256), 0, stream>>>(
            Wrel, Wroot, Wfc, Wq, Wk, Wv, Wo, Wmlp, wsB);
        encoder_node_kernel<1><<<dim3(B_ * N_), dim3(256), 0, stream>>>(
            X, Y, esrc, ew, pos, Wrel, brel, Wroot, Wfc, bfc,
            Wq, bq, Wk, bk, Wv, bv, Wo, bo, Wmlp, bmlp, wsB, out);
    } else {
        encoder_node_kernel<0><<<dim3(B_ * N_), dim3(256), 0, stream>>>(
            X, Y, esrc, ew, pos, Wrel, brel, Wroot, Wfc, bfc,
            Wq, bq, Wk, bk, Wv, bv, Wo, bo, Wmlp, bmlp, nullptr, out);
    }
}
